// GraphNetBlock_12043088297987
// MI455X (gfx1250) — matrix operand order repaired and checked by reading, not compiled, not run on hardware
//
#include <hip/hip_runtime.h>

typedef __attribute__((ext_vector_type(16))) _Float16 v16h;
typedef __attribute__((ext_vector_type(8)))  _Float16 v8h;
typedef __attribute__((ext_vector_type(4)))  _Float16 v4h;
typedef __attribute__((ext_vector_type(8)))  float    v8f;

#define DD 128
#define ROWS 128   // rows (edges/nodes) per block

// ---------- WMMA fragment loaders (layouts per CDNA5 ISA 7.12.2) ----------

__device__ __forceinline__ v16h cat8(v8h lo, v8h hi) {
  return __builtin_shufflevector(lo, hi, 0,1,2,3,4,5,6,7,8,9,10,11,12,13,14,15);
}

// A fragment: 16x32 f16 tile from LDS Xs[ROWS][128] row-major.
// slot j of lane L holds K = kt*32 + 8*(L>=16) + (j%8) + 16*(j/8), M = mtile*16 + (L%16)
__device__ __forceinline__ v16h load_a(const _Float16* Xs, int mtile, int kt, int lane) {
  int m    = mtile * 16 + (lane & 15);
  int base = m * DD + kt * 32 + ((lane >> 4) << 3);
  v8h lo = *(const v8h*)(Xs + base);        // K = kb .. kb+7
  v8h hi = *(const v8h*)(Xs + base + 16);   // K = kb+16 .. kb+23
  return cat8(lo, hi);
}

// B fragment: 32x16 f16 from global Wt[128][Ktot] row-major (pre-transposed weights).
// slot j of lane L holds K = kb + 8*(L>=16) + (j%8) + 16*(j/8), N = nb + (L%16)
__device__ __forceinline__ v16h load_b(const _Float16* __restrict__ Wt, int Ktot,
                                       int nb, int kb, int lane) {
  const _Float16* p = Wt + (size_t)(nb + (lane & 15)) * Ktot + kb + ((lane >> 4) << 3);
  v8h lo = *(const v8h*)p;
  v8h hi = *(const v8h*)(p + 16);
  return cat8(lo, hi);
}

// One 128-wide K segment: acc[p][nt] += A(mtile 2g+p) x B.  B fragment shared by both mtiles.
template <int KTILES>
__device__ __forceinline__ void gemm_acc(const _Float16* Xs, const _Float16* __restrict__ Wt,
                                         int Ktot, int ksegbase, int g, int nbase,
                                         int lane, v8f acc[2][4]) {
#pragma unroll
  for (int kt = 0; kt < KTILES; ++kt) {
    v16h a0 = load_a(Xs, 2 * g + 0, kt, lane);
    v16h a1 = load_a(Xs, 2 * g + 1, kt, lane);
#pragma unroll
    for (int nt = 0; nt < 4; ++nt) {
      v16h b = load_b(Wt, Ktot, nbase + nt * 16, ksegbase + kt * 32, lane);
      acc[0][nt] = __builtin_amdgcn_wmma_f32_16x16x32_f16(
          false, a0, false, b, (short)0, acc[0][nt], false, false);
      acc[1][nt] = __builtin_amdgcn_wmma_f32_16x16x32_f16(
          false, a1, false, b, (short)0, acc[1][nt], false, false);
    }
  }
}

// Epilogue: bias + ReLU, convert to f16, scatter into LDS H[ROWS][128]
__device__ __forceinline__ void epi_f16(const v8f acc[2][4], const float* __restrict__ bias,
                                        _Float16* Hs, int g, int nbase, int lane) {
#pragma unroll
  for (int p = 0; p < 2; ++p) {
#pragma unroll
    for (int nt = 0; nt < 4; ++nt) {
      int n = nbase + nt * 16 + (lane & 15);
      float bv = bias[n];
#pragma unroll
      for (int r = 0; r < 8; ++r) {
        int row = (2 * g + p) * 16 + r + ((lane >> 4) << 3);
        float v = acc[p][nt][r] + bv;
        v = v > 0.f ? v : 0.f;
        Hs[row * DD + n] = (_Float16)v;
      }
    }
  }
}

// Add bias into accumulators (pre-LayerNorm, layer 3)
__device__ __forceinline__ void add_bias(v8f acc[2][4], const float* __restrict__ bias,
                                         int nbase, int lane) {
#pragma unroll
  for (int nt = 0; nt < 4; ++nt) {
    float bv = bias[nbase + nt * 16 + (lane & 15)];
#pragma unroll
    for (int p = 0; p < 2; ++p)
#pragma unroll
      for (int r = 0; r < 8; ++r) acc[p][nt][r] += bv;
  }
}

// Per-row partial sums (64 cols in this wave) via xor-shuffle across the 16-lane group,
// written to LDS partials[row][nhalf].
__device__ __forceinline__ void ln_partials(const v8f acc[2][4], float* part_s, float* part_ss,
                                            int g, int nh, int lane) {
#pragma unroll
  for (int p = 0; p < 2; ++p) {
#pragma unroll
    for (int r = 0; r < 8; ++r) {
      float s = 0.f, ss = 0.f;
#pragma unroll
      for (int nt = 0; nt < 4; ++nt) {
        float v = acc[p][nt][r];
        s += v; ss += v * v;
      }
#pragma unroll
      for (int m = 1; m < 16; m <<= 1) {
        s  += __shfl_xor(s, m, 32);
        ss += __shfl_xor(ss, m, 32);
      }
      if ((lane & 15) == 0) {
        int row = (2 * g + p) * 16 + r + ((lane >> 4) << 3);
        part_s[row * 2 + nh]  = s;
        part_ss[row * 2 + nh] = ss;
      }
    }
  }
}

// ---------- prep kernels ----------

__global__ void k_zero4(float* __restrict__ p, long n4) {
  long i = (long)blockIdx.x * blockDim.x + threadIdx.x;
  if (i < n4) {
    float4 z; z.x = 0.f; z.y = 0.f; z.z = 0.f; z.w = 0.f;
    ((float4*)p)[i] = z;
  }
}

// W[K][128] f32 -> Wt[128][K] f16
__global__ void k_cvtT(const float* __restrict__ W, _Float16* __restrict__ Wt, int K) {
  int id = blockIdx.x * blockDim.x + threadIdx.x;
  if (id < K * DD) {
    int k = id >> 7;
    int n = id & (DD - 1);
    Wt[(size_t)n * K + k] = (_Float16)W[id];
  }
}

// ---------- edge kernel: gather -> 3-layer MLP (WMMA) -> LN -> out + atomic agg ----------

__global__ __launch_bounds__(256) void k_edge(
    const float* __restrict__ nodes, const float* __restrict__ edges,
    const int* __restrict__ snd, const int* __restrict__ rcv,
    const _Float16* __restrict__ W1t, const float* __restrict__ b1,
    const _Float16* __restrict__ W2t, const float* __restrict__ b2,
    const _Float16* __restrict__ W3t, const float* __restrict__ b3,
    const float* __restrict__ gamma, const float* __restrict__ beta,
    float* __restrict__ out_edges, float* __restrict__ agg, int Ne) {
  __shared__ __align__(16) char raw[65536];
  _Float16* Xs = (_Float16*)raw;             // 32 KB: K-segment staging, later H2
  _Float16* Hs = (_Float16*)(raw + 32768);   // 32 KB: H1
  // LN scratch overlays Xs after its last read (sync-protected):
  float* part_s  = (float*)raw;              // [128][2]
  float* part_ss = (float*)(raw + 1024);     // [128][2]
  float* s_mu    = (float*)(raw + 2048);     // [128]
  float* s_rs    = (float*)(raw + 2560);     // [128]

  const int tid   = threadIdx.x;
  const int lane  = tid & 31;
  const int wave  = tid >> 5;
  const int g     = wave >> 1;               // mtile pair 0..3 -> rows 32g..32g+31
  const int nh    = wave & 1;
  const int nbase = nh * 64;
  const long e0   = (long)blockIdx.x * ROWS;
  int evalid = Ne - (int)e0; if (evalid > ROWS) evalid = ROWS;

  const int r  = tid >> 1;                   // staging row 0..127 (2 threads/row)
  const int q  = tid & 1;
  const int rc = (r < evalid) ? r : (evalid - 1);
  const long er = e0 + rc;

  // ---- layer 1: K = 384 in three 128-wide segments ----
  v8f acc1[2][4] = {};
#pragma unroll
  for (int seg = 0; seg < 3; ++seg) {
    const float* src;
    if (seg == 0)      src = nodes + (size_t)snd[er] * DD;
    else if (seg == 1) src = nodes + (size_t)rcv[er] * DD;
    else               src = edges + (size_t)er * DD;
#pragma unroll
    for (int i = 0; i < 16; ++i) {
      int col = q * 64 + i * 4;
      float4 v = *(const float4*)(src + col);
      v4h h = {(_Float16)v.x, (_Float16)v.y, (_Float16)v.z, (_Float16)v.w};
      *(v4h*)(Xs + r * DD + col) = h;
    }
    __syncthreads();
    gemm_acc<4>(Xs, W1t, 384, seg * 128, g, nbase, lane, acc1);
    __syncthreads();
  }
  epi_f16(acc1, b1, Hs, g, nbase, lane);
  __syncthreads();

  // ---- layer 2: K = 128 ----
  v8f acc2[2][4] = {};
  gemm_acc<4>(Hs, W2t, 128, 0, g, nbase, lane, acc2);
  epi_f16(acc2, b2, Xs, g, nbase, lane);     // H2 -> Xs
  __syncthreads();

  // ---- layer 3: K = 128, stays in registers ----
  v8f acc3[2][4] = {};
  gemm_acc<4>(Xs, W3t, 128, 0, g, nbase, lane, acc3);
  add_bias(acc3, b3, nbase, lane);
  __syncthreads();                           // Xs dead -> LN scratch may overlay

  // ---- LayerNorm stats ----
  ln_partials(acc3, part_s, part_ss, g, nh, lane);
  __syncthreads();
  if (tid < ROWS) {
    float s  = part_s[tid * 2] + part_s[tid * 2 + 1];
    float ss = part_ss[tid * 2] + part_ss[tid * 2 + 1];
    float mu  = s * (1.f / DD);
    float var = ss * (1.f / DD) - mu * mu;
    s_mu[tid] = mu;
    s_rs[tid] = rsqrtf(var + 1e-5f);
  }
  __syncthreads();

  // ---- LN apply + residual + scatter-add, straight from accumulators ----
#pragma unroll
  for (int p = 0; p < 2; ++p) {
#pragma unroll
    for (int nt = 0; nt < 4; ++nt) {
      int col = nbase + nt * 16 + (lane & 15);
      float gm = gamma[col], bt = beta[col];
#pragma unroll
      for (int rr = 0; rr < 8; ++rr) {
        int row = (2 * g + p) * 16 + rr + ((lane >> 4) << 3);
        if (row < evalid) {
          long e = e0 + row;
          float v = (acc3[p][nt][rr] - s_mu[row]) * s_rs[row] * gm + bt;
          out_edges[e * DD + col] = v + edges[e * DD + col];          // residual
          __hip_atomic_fetch_add(&agg[(size_t)rcv[e] * DD + col], v, // pre-residual segment-sum
                                 __ATOMIC_RELAXED, __HIP_MEMORY_SCOPE_AGENT);
        }
      }
    }
  }
}

// ---------- node kernel: [nodes | agg] -> 3-layer MLP (WMMA) -> LN -> out ----------

__global__ __launch_bounds__(256) void k_node(
    const float* __restrict__ nodes, const float* __restrict__ agg,
    const _Float16* __restrict__ W1t, const float* __restrict__ b1,
    const _Float16* __restrict__ W2t, const float* __restrict__ b2,
    const _Float16* __restrict__ W3t, const float* __restrict__ b3,
    const float* __restrict__ gamma, const float* __restrict__ beta,
    float* __restrict__ out_nodes, int Nn) {
  __shared__ __align__(16) char raw[65536];
  _Float16* Xs = (_Float16*)raw;
  _Float16* Hs = (_Float16*)(raw + 32768);
  float* part_s  = (float*)raw;
  float* part_ss = (float*)(raw + 1024);
  float* s_mu    = (float*)(raw + 2048);
  float* s_rs    = (float*)(raw + 2560);

  const int tid   = threadIdx.x;
  const int lane  = tid & 31;
  const int wave  = tid >> 5;
  const int g     = wave >> 1;
  const int nh    = wave & 1;
  const int nbase = nh * 64;
  const long n0   = (long)blockIdx.x * ROWS;
  int nvalid = Nn - (int)n0; if (nvalid > ROWS) nvalid = ROWS;

  const int r  = tid >> 1;
  const int q  = tid & 1;
  const int rc = (r < nvalid) ? r : (nvalid - 1);
  const long nr = n0 + rc;

  // ---- layer 1: K = 256 in two 128-wide segments ----
  v8f acc1[2][4] = {};
#pragma unroll
  for (int seg = 0; seg < 2; ++seg) {
    const float* src = (seg == 0) ? (nodes + (size_t)nr * DD) : (agg + (size_t)nr * DD);
#pragma unroll
    for (int i = 0; i < 16; ++i) {
      int col = q * 64 + i * 4;
      float4 v = *(const float4*)(src + col);
      v4h h = {(_Float16)v.x, (_Float16)v.y, (_Float16)v.z, (_Float16)v.w};
      *(v4h*)(Xs + r * DD + col) = h;
    }
    __syncthreads();
    gemm_acc<4>(Xs, W1t, 256, seg * 128, g, nbase, lane, acc1);
    __syncthreads();
  }
  epi_f16(acc1, b1, Hs, g, nbase, lane);
  __syncthreads();

  v8f acc2[2][4] = {};
  gemm_acc<4>(Hs, W2t, 128, 0, g, nbase, lane, acc2);
  epi_f16(acc2, b2, Xs, g, nbase, lane);
  __syncthreads();

  v8f acc3[2][4] = {};
  gemm_acc<4>(Xs, W3t, 128, 0, g, nbase, lane, acc3);
  add_bias(acc3, b3, nbase, lane);
  __syncthreads();

  ln_partials(acc3, part_s, part_ss, g, nh, lane);
  __syncthreads();
  if (tid < ROWS) {
    float s  = part_s[tid * 2] + part_s[tid * 2 + 1];
    float ss = part_ss[tid * 2] + part_ss[tid * 2 + 1];
    float mu  = s * (1.f / DD);
    float var = ss * (1.f / DD) - mu * mu;
    s_mu[tid] = mu;
    s_rs[tid] = rsqrtf(var + 1e-5f);
  }
  __syncthreads();

#pragma unroll
  for (int p = 0; p < 2; ++p) {
#pragma unroll
    for (int nt = 0; nt < 4; ++nt) {
      int col = nbase + nt * 16 + (lane & 15);
      float gm = gamma[col], bt = beta[col];
#pragma unroll
      for (int rr = 0; rr < 8; ++rr) {
        int row = (2 * g + p) * 16 + rr + ((lane >> 4) << 3);
        if (row < nvalid) {
          long n = n0 + row;
          float v = (acc3[p][nt][rr] - s_mu[row]) * s_rs[row] * gm + bt;
          out_nodes[n * DD + col] = v + nodes[n * DD + col];         // residual
        }
      }
    }
  }
}

// ---------- launch ----------

extern "C" void kernel_launch(void* const* d_in, const int* in_sizes, int n_in,
                              void* d_out, int out_size, void* d_ws, size_t ws_size,
                              hipStream_t stream) {
  const float* nodes = (const float*)d_in[0];
  const float* edges = (const float*)d_in[1];
  const int*   snd   = (const int*)d_in[2];
  const int*   rcv   = (const int*)d_in[3];
  const float* eW1 = (const float*)d_in[4];
  const float* eb1 = (const float*)d_in[5];
  const float* eW2 = (const float*)d_in[6];
  const float* eb2 = (const float*)d_in[7];
  const float* eW3 = (const float*)d_in[8];
  const float* eb3 = (const float*)d_in[9];
  const float* eg  = (const float*)d_in[10];
  const float* ebt = (const float*)d_in[11];
  const float* nW1 = (const float*)d_in[12];
  const float* nb1 = (const float*)d_in[13];
  const float* nW2 = (const float*)d_in[14];
  const float* nb2 = (const float*)d_in[15];
  const float* nW3 = (const float*)d_in[16];
  const float* nb3 = (const float*)d_in[17];
  const float* ng  = (const float*)d_in[18];
  const float* nbt = (const float*)d_in[19];

  const int Nn = in_sizes[0] / DD;   // 100000
  const int Ne = in_sizes[2];        // 400000

  float* out_nodes = (float*)d_out;
  float* out_edges = (float*)d_out + (size_t)Nn * DD;

  // workspace: agg[N,128] f32, then 6 transposed f16 weight copies
  char* w = (char*)d_ws;
  float* agg = (float*)w;
  size_t off = (((size_t)Nn * DD * sizeof(float)) + 255) & ~(size_t)255;
  _Float16* WtE1 = (_Float16*)(w + off); off += (size_t)DD * 384 * 2;
  _Float16* WtE2 = (_Float16*)(w + off); off += (size_t)DD * 128 * 2;
  _Float16* WtE3 = (_Float16*)(w + off); off += (size_t)DD * 128 * 2;
  _Float16* WtN1 = (_Float16*)(w + off); off += (size_t)DD * 256 * 2;
  _Float16* WtN2 = (_Float16*)(w + off); off += (size_t)DD * 128 * 2;
  _Float16* WtN3 = (_Float16*)(w + off); off += (size_t)DD * 128 * 2;

  long n4 = (long)Nn * DD / 4;
  k_zero4<<<(int)((n4 + 255) / 256), 256, 0, stream>>>(agg, n4);
  k_cvtT<<<(384 * DD + 255) / 256, 256, 0, stream>>>(eW1, WtE1, 384);
  k_cvtT<<<(128 * DD + 255) / 256, 256, 0, stream>>>(eW2, WtE2, 128);
  k_cvtT<<<(128 * DD + 255) / 256, 256, 0, stream>>>(eW3, WtE3, 128);
  k_cvtT<<<(256 * DD + 255) / 256, 256, 0, stream>>>(nW1, WtN1, 256);
  k_cvtT<<<(128 * DD + 255) / 256, 256, 0, stream>>>(nW2, WtN2, 128);
  k_cvtT<<<(128 * DD + 255) / 256, 256, 0, stream>>>(nW3, WtN3, 128);

  k_edge<<<(Ne + ROWS - 1) / ROWS, 256, 0, stream>>>(nodes, edges, snd, rcv,
                                                     WtE1, eb1, WtE2, eb2, WtE3, eb3,
                                                     eg, ebt, out_edges, agg, Ne);
  k_node<<<(Nn + ROWS - 1) / ROWS, 256, 0, stream>>>(nodes, agg,
                                                     WtN1, nb1, WtN2, nb2, WtN3, nb3,
                                                     ng, nbt, out_nodes, Nn);
}
